// WaveNet_16965120819940
// MI455X (gfx1250) — matrix operand order repaired and checked by reading, not compiled, not run on hardware
//
#include <hip/hip_runtime.h>

#define NLAYERS 30
#define RDIM 64
#define SDIM 256
#define UDIM 256
#define TLEN 16384
#define BSZ 8

typedef __attribute__((ext_vector_type(16))) __bf16 v16bf;
typedef __attribute__((ext_vector_type(8)))  float  v8f;
typedef __attribute__((ext_vector_type(4)))  int    v4i;
typedef __attribute__((address_space(1))) v4i gv4i;   // global v4i
typedef __attribute__((address_space(3))) v4i lv4i;   // LDS v4i

union AF { v16bf v; __bf16 e[16]; uint4 q[2]; };

__device__ __forceinline__ v8f wmma_bf(v16bf a, v16bf b, v8f c) {
  // D = A(16x32 bf16) * B(32x16 bf16) + C(16x16 f32)
  return __builtin_amdgcn_wmma_f32_16x16x32_bf16(false, a, false, b, (short)0, c, false, false);
}

__device__ __forceinline__ v16bf pk_frag(const __bf16* p) {
  AF f;
  f.q[0] = *(const uint4*)(p);
  f.q[1] = *(const uint4*)(p + 8);
  return f.v;
}

__device__ __forceinline__ v16bf a_zero() {
  AF f;
  f.q[0] = make_uint4(0u, 0u, 0u, 0u);
  f.q[1] = make_uint4(0u, 0u, 0u, 0u);
  return f.v;
}

__device__ __forceinline__ void cvt4(AF& f, int base, float4 u) {
  f.e[base+0] = (__bf16)u.x; f.e[base+1] = (__bf16)u.y;
  f.e[base+2] = (__bf16)u.z; f.e[base+3] = (__bf16)u.w;
}

// A-fragment from an f32 row: elements 0..7 <- K=k0..k0+7, 8..15 <- K=k1..k1+7
__device__ __forceinline__ v16bf a_f32(const float* rowp, int k0, int k1) {
  AF f;
  cvt4(f, 0,  *(const float4*)(rowp + k0));
  cvt4(f, 4,  *(const float4*)(rowp + k0 + 4));
  cvt4(f, 8,  *(const float4*)(rowp + k1));
  cvt4(f, 12, *(const float4*)(rowp + k1 + 4));
  return f.v;
}

__device__ __forceinline__ void cvt4r(AF& f, int base, float4 u, float4 b) {
  f.e[base+0] = (__bf16)fmaxf(u.x + b.x, 0.f); f.e[base+1] = (__bf16)fmaxf(u.y + b.y, 0.f);
  f.e[base+2] = (__bf16)fmaxf(u.z + b.z, 0.f); f.e[base+3] = (__bf16)fmaxf(u.w + b.w, 0.f);
}

// A-fragment = relu(row + bias) in bf16
__device__ __forceinline__ v16bf a_relu_f32(const float* rowp, const float* bp, int k0, int k1) {
  AF f;
  cvt4r(f, 0,  *(const float4*)(rowp + k0),     *(const float4*)(bp + k0));
  cvt4r(f, 4,  *(const float4*)(rowp + k0 + 4), *(const float4*)(bp + k0 + 4));
  cvt4r(f, 8,  *(const float4*)(rowp + k1),     *(const float4*)(bp + k1));
  cvt4r(f, 12, *(const float4*)(rowp + k1 + 4), *(const float4*)(bp + k1 + 4));
  return f.v;
}

__device__ __forceinline__ v16bf a_bf16(const __bf16* rowp, int k0, int k1) {
  AF f;
  f.q[0] = *(const uint4*)(rowp + k0);
  f.q[1] = *(const uint4*)(rowp + k1);
  return f.v;
}

__device__ __forceinline__ float sigm(float x) { return 1.f / (1.f + __expf(-x)); }

// ---------------------------------------------------------------------------
// Async Global -> LDS bulk staging (CDNA5 GLOBAL_LOAD_ASYNC_TO_LDS_B128,
// tracked with ASYNCcnt). Builtin signature: (global v4i*, lds v4i*, imm, cpol)
// ---------------------------------------------------------------------------
__device__ __forceinline__ void async_b128(__bf16* lds_dst, const __bf16* gsrc) {
#if __has_builtin(__builtin_amdgcn_global_load_async_to_lds_b128)
  __builtin_amdgcn_global_load_async_to_lds_b128(
      (gv4i*)gsrc, (lv4i*)lds_dst, 0, 0);
#else
  unsigned loff = (unsigned)(unsigned long long)
      (__attribute__((address_space(3))) __bf16*)lds_dst;
  asm volatile("global_load_async_to_lds_b128 %0, %1, off"
               :: "v"(loff), "v"(gsrc) : "memory");
#endif
}

__device__ __forceinline__ void wait_async0() {
#if __has_builtin(__builtin_amdgcn_s_wait_asynccnt)
  __builtin_amdgcn_s_wait_asynccnt(0);
#else
  asm volatile("s_wait_asynccnt 0x0" ::: "memory");
#endif
}

// copy n_elems bf16 (n_elems % 1024 == 0) global -> LDS, 128 threads, 16B units
__device__ __forceinline__ void stage_region(const __bf16* g, __bf16* l,
                                             int n_elems, int tid) {
  for (int i = tid * 8; i < n_elems; i += 128 * 8)
    async_b128(l + i, g + i);
}

// ---------------------------------------------------------------------------
// Pack an f32 [K x N] matrix (row-major) into bf16 B-matrix fragments:
// layout [ntile][kchunk][lane][16]; lane<16: N=nt*16+lane, K=kc*32+{0..7,16..23}
//                                  lane>=16: N=nt*16+lane-16, K=kc*32+{8..15,24..31}
// ---------------------------------------------------------------------------
__global__ void pack_b_kernel(const float* __restrict__ src, __bf16* __restrict__ dst,
                              int K, int N, int mats) {
  const int KC = K / 32, NT = N / 16;
  const int per_mat = NT * KC * 32;
  const int tid = blockIdx.x * blockDim.x + threadIdx.x;
  if (tid >= mats * per_mat) return;
  const int m    = tid / per_mat;
  const int rem  = tid % per_mat;
  const int nt   = rem / (KC * 32);
  const int rem2 = rem % (KC * 32);
  const int kc   = rem2 / 32;
  const int lane = rem2 % 32;
  const float* s = src + (size_t)m * K * N;
  __bf16* dp = dst + (((size_t)m * NT + nt) * KC + kc) * 512 + lane * 16;
  const int kbase = kc * 32 + (lane >> 4) * 8;
  const int n = nt * 16 + (lane & 15);
  for (int e = 0; e < 16; ++e)
    dp[e] = (__bf16)s[(size_t)(kbase + (e < 8 ? e : e + 8)) * N + n];
}

// bias[k][b][n] = conv_b[k][n] + sum_r sid_table[sid[b]][r] * proj_w[k][r][n]
__global__ void bias_kernel(const float* __restrict__ conv_b, const float* __restrict__ proj_w,
                            const float* __restrict__ sid_table, const int* __restrict__ sid,
                            float* __restrict__ bias) {
  const int tid = blockIdx.x * blockDim.x + threadIdx.x;
  if (tid >= NLAYERS * BSZ * 128) return;
  const int n = tid & 127;
  const int b = (tid >> 7) & 7;
  const int k = tid >> 10;
  const float* iv = sid_table + (size_t)sid[b] * RDIM;
  const float* pw = proj_w + (size_t)k * RDIM * 128;
  float acc = conv_b[k * 128 + n];
  for (int r = 0; r < RDIM; ++r) acc += iv[r] * pw[r * 128 + n];
  bias[((size_t)k * BSZ + b) * 128 + n] = acc;
}

// ---------------------------------------------------------------------------
// Embed: x = tanh(conv(oh, ew, k=2, d=1) + eb)   [B,T,U=256] -> [B,T,R=64]
// ew fragments (64KB) staged in LDS via async copy.
// ---------------------------------------------------------------------------
__global__ void __launch_bounds__(128) embed_kernel(
    const float* __restrict__ oh, const float* __restrict__ eb,
    const __bf16* __restrict__ ew_pk, float* __restrict__ x) {
  __shared__ __align__(16) __bf16 estage[32768];  // [tap0 16384][tap1 16384]
  const int tid = threadIdx.x, lane = tid & 31, wave = tid >> 5;
  const int ln = lane & 15, hi = (lane >> 4) * 8;

  stage_region(ew_pk, estage, 32768, tid);

  const int gt = blockIdx.x * 4 + wave;
  const int bt0 = gt * 16;
  const int b = bt0 / TLEN, t0 = bt0 % TLEN;
  const long bT0 = (long)b * TLEN + t0;
  const int t = t0 + ln;
  const float* rowc = oh + ((long)b * TLEN + t) * UDIM;
  const float* rowp = rowc - UDIM;   // t-1
  const bool havep = (t > 0);

  wait_async0();
  __syncthreads();
  const __bf16* w0 = estage;           // tap 0 (t-1), NT=4 KC=8
  const __bf16* w1 = estage + 16384;   // tap 1 (t)

  v8f vzero = {};
  v8f acc[4];
#pragma unroll
  for (int nt = 0; nt < 4; ++nt) acc[nt] = vzero;

  for (int c = 0; c < 8; ++c) {
    const int cb = c * 32;
    v16bf acu = a_f32(rowc, cb + hi, cb + 16 + hi);
    v16bf apr = havep ? a_f32(rowp, cb + hi, cb + 16 + hi) : a_zero();
#pragma unroll
    for (int nt = 0; nt < 4; ++nt) {
      v16bf b0 = pk_frag(w0 + (size_t)(nt * 8 + c) * 512 + lane * 16);
      v16bf b1 = pk_frag(w1 + (size_t)(nt * 8 + c) * 512 + lane * 16);
      acc[nt] = wmma_bf(apr, b0, acc[nt]);
      acc[nt] = wmma_bf(acu, b1, acc[nt]);
    }
  }
#pragma unroll
  for (int nt = 0; nt < 4; ++nt) {
    const int N = nt * 16 + ln;
    const float bb = eb[N];
#pragma unroll
    for (int j = 0; j < 8; ++j)
      x[(bT0 + j + hi) * RDIM + N] = tanhf(acc[nt][j] + bb);
  }
}

// ---------------------------------------------------------------------------
// One WaveNet layer: dilated conv -> gate -> residual + skip accumulation.
// Per-layer packed weights (72KB) staged into LDS with async copy; B frags
// then come from LDS (ds_load_b128) instead of per-wave global loads.
// ---------------------------------------------------------------------------
__global__ void __launch_bounds__(128) layer_kernel(
    const float* __restrict__ xin, float* __restrict__ xout,
    float* __restrict__ skip,
    const __bf16* __restrict__ conv_pk, const __bf16* __restrict__ one_pk,
    const __bf16* __restrict__ skip_pk,
    const float* __restrict__ bias, const float* __restrict__ one_b,
    int layer, int dil, int first) {
  __shared__ __align__(16) __bf16 wstage[36864]; // [w0 8192][w1 8192][one 4096][skip 16384]
  __shared__ __align__(16) __bf16 zl[4][16][RDIM];
  const int tid = threadIdx.x, lane = tid & 31, wave = tid >> 5;
  const int ln = lane & 15, hi = (lane >> 4) * 8;

  // kick off async weight staging first, overlap with A-fragment loads
  stage_region(conv_pk + (size_t)layer * 16384, wstage,         16384, tid);
  stage_region(one_pk  + (size_t)layer * 4096,  wstage + 16384,  4096, tid);
  stage_region(skip_pk + (size_t)layer * 16384, wstage + 20480, 16384, tid);

  const int gt = blockIdx.x * 4 + wave;
  const int bt0 = gt * 16;
  const int b = bt0 / TLEN, t0 = bt0 % TLEN;
  const long bT0 = (long)b * TLEN + t0;

  // A fragments: x[t] and x[t-d] (causal zero pad)
  const int t = t0 + ln;
  const float* rowc = xin + ((long)b * TLEN + t) * RDIM;
  v16bf ac0 = a_f32(rowc, 0 + hi, 16 + hi);
  v16bf ac1 = a_f32(rowc, 32 + hi, 48 + hi);
  v16bf ap0, ap1;
  const int tp = t - dil;
  if (tp >= 0) {
    const float* rowp = xin + ((long)b * TLEN + tp) * RDIM;
    ap0 = a_f32(rowp, 0 + hi, 16 + hi);
    ap1 = a_f32(rowp, 32 + hi, 48 + hi);
  } else {
    ap0 = a_zero(); ap1 = a_zero();
  }

  wait_async0();
  __syncthreads();
  const __bf16* w0 = wstage;           // conv tap 0, NT=8 KC=2
  const __bf16* w1 = wstage + 8192;    // conv tap 1
  const __bf16* ow = wstage + 16384;   // one_w, NT=4 KC=2
  const __bf16* sw = wstage + 20480;   // skip_w, NT=16 KC=2

  // dilated conv: Wx = x_prev@W0 + x_cur@W1  (16 x 128)
  v8f vzero = {};
  v8f wacc[8];
#pragma unroll
  for (int nt = 0; nt < 8; ++nt) {
    v16bf b0 = pk_frag(w0 + (size_t)(nt * 2 + 0) * 512 + lane * 16);
    v16bf b1 = pk_frag(w0 + (size_t)(nt * 2 + 1) * 512 + lane * 16);
    v16bf b2 = pk_frag(w1 + (size_t)(nt * 2 + 0) * 512 + lane * 16);
    v16bf b3 = pk_frag(w1 + (size_t)(nt * 2 + 1) * 512 + lane * 16);
    v8f a = vzero;
    a = wmma_bf(ap0, b0, a);
    a = wmma_bf(ap1, b1, a);
    a = wmma_bf(ac0, b2, a);
    a = wmma_bf(ac1, b3, a);
    wacc[nt] = a;
  }

  // gate: z = tanh(sm[:R]) * sigmoid(sm[R:]); write bf16 to LDS (C->A relayout)
  const float* bs = bias + ((size_t)layer * BSZ + b) * 128;
#pragma unroll
  for (int nt = 0; nt < 4; ++nt) {
    const float bt_ = bs[nt * 16 + ln];
    const float bg  = bs[64 + nt * 16 + ln];
#pragma unroll
    for (int j = 0; j < 8; ++j) {
      const float tv = tanhf(wacc[nt][j] + bt_);
      const float gv = sigm(wacc[nt + 4][j] + bg);
      zl[wave][j + hi][nt * 16 + ln] = (__bf16)(tv * gv);
    }
  }
  __syncthreads();

  const __bf16* zrow = &zl[wave][ln][0];
  v16bf za0 = a_bf16(zrow, 0 + hi, 16 + hi);
  v16bf za1 = a_bf16(zrow, 32 + hi, 48 + hi);

  // residual: x_out = z @ one_w + one_b + x_in
#pragma unroll
  for (int nt = 0; nt < 4; ++nt) {
    v16bf b0 = pk_frag(ow + (size_t)(nt * 2 + 0) * 512 + lane * 16);
    v16bf b1 = pk_frag(ow + (size_t)(nt * 2 + 1) * 512 + lane * 16);
    v8f a = vzero;
    a = wmma_bf(za0, b0, a);
    a = wmma_bf(za1, b1, a);
    const int N = nt * 16 + ln;
    const float ob = one_b[layer * RDIM + N];
#pragma unroll
    for (int j = 0; j < 8; ++j) {
      const long idx = (bT0 + j + hi) * RDIM + N;
      xout[idx] = a[j] + ob + xin[idx];
    }
  }

  // skip accumulation: skip += z @ skip_w
  for (int nt = 0; nt < 16; ++nt) {
    v16bf b0 = pk_frag(sw + (size_t)(nt * 2 + 0) * 512 + lane * 16);
    v16bf b1 = pk_frag(sw + (size_t)(nt * 2 + 1) * 512 + lane * 16);
    v8f a = vzero;
    a = wmma_bf(za0, b0, a);
    a = wmma_bf(za1, b1, a);
    const int N = nt * 16 + ln;
#pragma unroll
    for (int j = 0; j < 8; ++j) {
      const long idx = (bT0 + j + hi) * SDIM + N;
      if (first) skip[idx] = a[j];
      else       skip[idx] += a[j];
    }
  }
}

// ---------------------------------------------------------------------------
// Head: out = relu(relu(skip+sb) @ d1_w + d1_b) @ d2_w + d2_b
// d1/d2 fragment tables are 128KB each -> left in L2 (fully resident).
// ---------------------------------------------------------------------------
__global__ void __launch_bounds__(128) head_kernel(
    const float* __restrict__ skip, const float* __restrict__ sb,
    const __bf16* __restrict__ d1_pk, const float* __restrict__ d1_b,
    const __bf16* __restrict__ d2_pk, const float* __restrict__ d2_b,
    float* __restrict__ out) {
  __shared__ __align__(16) __bf16 s1[4][16][UDIM];
  const int tid = threadIdx.x, lane = tid & 31, wave = tid >> 5;
  const int ln = lane & 15, hi = (lane >> 4) * 8;
  const int gt = blockIdx.x * 4 + wave;
  const int bt0 = gt * 16;
  const int b = bt0 / TLEN, t0 = bt0 % TLEN;
  const long bT0 = (long)b * TLEN + t0;
  const int t = t0 + ln;
  const float* row = skip + ((long)b * TLEN + t) * SDIM;

  v8f vzero = {};
  v8f acc[16];
#pragma unroll
  for (int nt = 0; nt < 16; ++nt) acc[nt] = vzero;

  // s1 = relu( relu(skip+sb) @ d1_w + d1_b ), NT=16 KC=8
  for (int c = 0; c < 8; ++c) {
    const int cb = c * 32;
    v16bf av = a_relu_f32(row, sb, cb + hi, cb + 16 + hi);
#pragma unroll
    for (int nt = 0; nt < 16; ++nt)
      acc[nt] = wmma_bf(av, pk_frag(d1_pk + (size_t)(nt * 8 + c) * 512 + lane * 16), acc[nt]);
  }
#pragma unroll
  for (int nt = 0; nt < 16; ++nt) {
    const float bb = d1_b[nt * 16 + ln];
#pragma unroll
    for (int j = 0; j < 8; ++j) {
      const float v = acc[nt][j] + bb;
      s1[wave][j + hi][nt * 16 + ln] = (__bf16)(v > 0.f ? v : 0.f);
    }
  }
  __syncthreads();

  // out = s1 @ d2_w + d2_b
#pragma unroll
  for (int nt = 0; nt < 16; ++nt) acc[nt] = vzero;
  const __bf16* zrow = &s1[wave][ln][0];
  for (int c = 0; c < 8; ++c) {
    const int cb = c * 32;
    v16bf av = a_bf16(zrow, cb + hi, cb + 16 + hi);
#pragma unroll
    for (int nt = 0; nt < 16; ++nt)
      acc[nt] = wmma_bf(av, pk_frag(d2_pk + (size_t)(nt * 8 + c) * 512 + lane * 16), acc[nt]);
  }
#pragma unroll
  for (int nt = 0; nt < 16; ++nt) {
    const int N = nt * 16 + ln;
    const float bb = d2_b[N];
#pragma unroll
    for (int j = 0; j < 8; ++j)
      out[(bT0 + j + hi) * UDIM + N] = acc[nt][j] + bb;
  }
}

// ---------------------------------------------------------------------------
extern "C" void kernel_launch(void* const* d_in, const int* in_sizes, int n_in,
                              void* d_out, int out_size, void* d_ws, size_t ws_size,
                              hipStream_t stream) {
  (void)in_sizes; (void)n_in; (void)out_size; (void)ws_size;
  const float* oh        = (const float*)d_in[0];
  const int*   sid       = (const int*)  d_in[1];
  const float* ew        = (const float*)d_in[2];
  const float* eb        = (const float*)d_in[3];
  const float* sid_table = (const float*)d_in[4];
  const float* conv_w    = (const float*)d_in[5];
  const float* conv_b    = (const float*)d_in[6];
  const float* proj_w    = (const float*)d_in[7];
  const float* one_w     = (const float*)d_in[8];
  const float* one_b     = (const float*)d_in[9];
  const float* skip_w    = (const float*)d_in[10];
  const float* sb        = (const float*)d_in[11];
  const float* d1_w      = (const float*)d_in[12];
  const float* d1_b      = (const float*)d_in[13];
  const float* d2_w      = (const float*)d_in[14];
  const float* d2_b      = (const float*)d_in[15];
  float* out = (float*)d_out;

  // workspace carve (~195 MB): xA/xB ping-pong + skip stay L2-resident (192MB L2)
  char* p = (char*)d_ws;
  auto carve = [&](size_t bytes) -> void* {
    void* r = (void*)p;
    p += (bytes + 255) & ~(size_t)255;
    return r;
  };
  float*  xA      = (float*)carve((size_t)BSZ * TLEN * RDIM * 4);
  float*  xB      = (float*)carve((size_t)BSZ * TLEN * RDIM * 4);
  float*  skip    = (float*)carve((size_t)BSZ * TLEN * SDIM * 4);
  float*  bias    = (float*)carve((size_t)NLAYERS * BSZ * 128 * 4);
  __bf16* ew_pk   = (__bf16*)carve((size_t)2 * 16384 * 2);       // NT=4 KC=8
  __bf16* conv_pk = (__bf16*)carve((size_t)60 * 8192 * 2);       // NT=8 KC=2
  __bf16* one_pk  = (__bf16*)carve((size_t)30 * 4096 * 2);       // NT=4 KC=2
  __bf16* skip_pk = (__bf16*)carve((size_t)30 * 16384 * 2);      // NT=16 KC=2
  __bf16* d1_pk   = (__bf16*)carve((size_t)65536 * 2);           // NT=16 KC=8
  __bf16* d2_pk   = (__bf16*)carve((size_t)65536 * 2);

  // weight packing + conditioning bias (tiny, once per call)
  pack_b_kernel<<<dim3(8),   dim3(256), 0, stream>>>(ew,     ew_pk,   256, 64,  2);
  pack_b_kernel<<<dim3(120), dim3(256), 0, stream>>>(conv_w, conv_pk, 64,  128, 60);
  pack_b_kernel<<<dim3(30),  dim3(256), 0, stream>>>(one_w,  one_pk,  64,  64,  30);
  pack_b_kernel<<<dim3(120), dim3(256), 0, stream>>>(skip_w, skip_pk, 64,  256, 30);
  pack_b_kernel<<<dim3(16),  dim3(256), 0, stream>>>(d1_w,   d1_pk,   256, 256, 1);
  pack_b_kernel<<<dim3(16),  dim3(256), 0, stream>>>(d2_w,   d2_pk,   256, 256, 1);
  bias_kernel<<<dim3(120), dim3(256), 0, stream>>>(conv_b, proj_w, sid_table, sid, bias);

  const int nblk = (BSZ * TLEN) / 64;  // 4 waves x 16 timesteps per block
  embed_kernel<<<dim3(nblk), dim3(128), 0, stream>>>(oh, eb, ew_pk, xA);

  for (int k = 0; k < NLAYERS; ++k) {
    const int d = 1 << (k % 10);
    const float* xi = (k & 1) ? xB : xA;
    float*       xo = (k & 1) ? xA : xB;
    layer_kernel<<<dim3(nblk), dim3(128), 0, stream>>>(
        xi, xo, skip, conv_pk, one_pk, skip_pk, bias, one_b, k, d, (k == 0) ? 1 : 0);
  }

  head_kernel<<<dim3(nblk), dim3(128), 0, stream>>>(skip, sb, d1_pk, d1_b, d2_pk, d2_b, out);
}
